// VectorQuantizer_67723044323837
// MI455X (gfx1250) — matrix operand order repaired and checked by reading, not compiled, not run on hardware
//
#include <hip/hip_runtime.h>
#include <hip/hip_bf16.h>

// CDNA5 (gfx1250) vector types for WMMA
typedef __attribute__((ext_vector_type(16))) __bf16 v16bf;
typedef __attribute__((ext_vector_type(8)))  float  v8f;
// Native clang vector for async-LDS builtin (must be vector_size, not HIP int4)
typedef int v4i __attribute__((vector_size(16)));

// Problem constants: z [64,256,32,32] f32, emb [1024,256] f32
#define NB      64
#define NC      256          // channels (GEMM K dim)
#define HW      1024         // 32*32 tokens per batch image
#define NTOK    65536        // 64*1024 tokens
#define NK      1024         // codebook entries (GEMM N dim)
#define NELEM   16777216     // NTOK * NC
#define BETA    0.25f

// Workspace layout (bytes):
//   [0, 524288)          : emb in bf16, row-major [1024][256]
//   [524288, 528384)     : e_sq[1024] f32
//   [528384, 530432)     : per-block loss partials, 256 x float2
#define WS_EBF_OFF   0
#define WS_ESQ_OFF   524288
#define WS_PART_OFF  528384

// Async global->LDS staging (gfx1250 ASYNCcnt path) if toolchain exposes it.
// Diagnosed signature: (v4i __device__* src, v4i __shared__* dst,
//                       imm int offset, imm int cpol)
#if defined(__has_builtin)
#if __has_builtin(__builtin_amdgcn_global_load_async_to_lds_b128) && \
    __has_builtin(__builtin_amdgcn_s_wait_asynccnt)
#define USE_ASYNC_LDS 1
#endif
#endif

// Copy one 8KB B slab (16 codes x 256 ch bf16, contiguous) into LDS.
__device__ __forceinline__ void stage_slab(const __bf16* __restrict__ src,
                                           __bf16* __restrict__ dst, int tid) {
#if defined(USE_ASYNC_LDS)
    v4i* gsrc = (v4i*)src;             // drops const; implicit AS conversion
    v4i* ldst = (v4i*)dst;
    __builtin_amdgcn_global_load_async_to_lds_b128(gsrc + tid,       ldst + tid,       0, 0);
    __builtin_amdgcn_global_load_async_to_lds_b128(gsrc + tid + 256, ldst + tid + 256, 0, 0);
#else
    const v4i* gsrc = (const v4i*)src;
    v4i* ldst = (v4i*)dst;
    ldst[tid]       = gsrc[tid];
    ldst[tid + 256] = gsrc[tid + 256];
#endif
}

__device__ __forceinline__ void stage_fence() {
#if defined(USE_ASYNC_LDS)
    __builtin_amdgcn_s_wait_asynccnt(0);
#endif
    __syncthreads();
}

// ---------------------------------------------------------------------------
// Kernel 1: convert codebook to bf16 (row major == B-fragment feed layout)
// and compute exact f32 squared norms e_sq[k].
// ---------------------------------------------------------------------------
__global__ void vq_prep_emb(const float* __restrict__ emb,
                            __bf16* __restrict__ ebf,
                            float* __restrict__ esq) {
    const int k   = blockIdx.x;       // code row, 0..1023
    const int tid = threadIdx.x;      // 0..255 (channel)
    const float v = emb[k * NC + tid];
    ebf[k * NC + tid] = (__bf16)v;    // RNE convert
    float s = v * v;
    #pragma unroll
    for (int o = 16; o; o >>= 1) s += __shfl_xor(s, o, 32);
    __shared__ float r8[8];
    if ((tid & 31) == 0) r8[tid >> 5] = s;
    __syncthreads();
    if (tid == 0) {
        float t = 0.f;
        #pragma unroll
        for (int w = 0; w < 8; ++w) t += r8[w];
        esq[k] = t;
    }
}

// ---------------------------------------------------------------------------
// Kernel 2: distance GEMM via v_wmma_f32_16x16x32_bf16, double-buffered async
// LDS B staging, 2 token tiles per wave (each B fragment feeds 2 WMMAs),
// running argmin, gather/scatter epilogue, deterministic loss partials.
// Grid: 256 blocks x 256 threads (8 wave32). Wave owns 32 tokens (2 tiles).
// ---------------------------------------------------------------------------
__global__ void __launch_bounds__(256)
vq_main(const float* __restrict__ zin,
        const float* __restrict__ emb,
        const __bf16* __restrict__ ebf,
        const float* __restrict__ esq,
        float* __restrict__ out,
        float2* __restrict__ partials) {
    // Ping-pong B buffers: 2 x (16 codes x 256 ch bf16) = 16 KB
    __shared__ __align__(32) __bf16 ldsB[2][16 * NC];
    __shared__ float rs[8], rd[8];

    const int tid  = threadIdx.x;
    const int lane = tid & 31;
    const int wave = tid >> 5;
    const int m    = lane & 15;          // row (A) / column (B,C) within tile
    const int hi   = lane >> 4;          // half-wave select
    const int g    = blockIdx.x;         // 256 blocks, 256 tokens each

    const int b      = g >> 2;                  // batch image (256 | 1024)
    const int hwBase = (g & 3) * 256;           // hw offset of this WG's tokens
    const int hwA0   = hwBase + wave * 32 + m;  // tile0 row token hw
    // tile1 rows are hwA0 + 16

    // ---- Load two A tiles (32 tokens x 256 ch) as bf16 WMMA fragments -----
    // A fragments use contiguous k as chosen here (B matches): lane holds row M=lane%16;
// half 0 (lanes 0-15): K = {0..15}; half 1: K = {16..31}.
    const int ka = hi * 16;
    const float* zA = zin + ((size_t)b * NC) * (size_t)HW + (size_t)hwA0;
    union AFrag { v16bf v; __bf16 e[16]; };
    AFrag a0[8], a1[8];
    #pragma unroll
    for (int kc = 0; kc < 8; ++kc) {
        #pragma unroll
        for (int e = 0; e < 16; ++e) {
            const int k  = ka + e;
            const int ch = kc * 32 + k;
            a0[kc].e[e] = (__bf16)zA[(size_t)ch << 10];
            a1[kc].e[e] = (__bf16)zA[((size_t)ch << 10) + 16];
        }
    }

    // ---- Running per-row (min, argmin), 8 rows per tile per lane ----------
    float best0[8], best1[8];
    int   bidx0[8], bidx1[8];
    #pragma unroll
    for (int r = 0; r < 8; ++r) {
        best0[r] = 3.4e38f; bidx0[r] = 0;
        best1[r] = 3.4e38f; bidx1[r] = 0;
    }

    const int kb16 = hi * 16;            // B fragment K base for this half

    stage_slab(ebf, ldsB[0], tid);       // preload first slab
    stage_fence();

    for (int nc = 0; nc < 64; ++nc) {
        const int cur = nc & 1;
        if (nc + 1 < 64)                  // overlap next slab with compute
            stage_slab(ebf + (size_t)(nc + 1) * 4096, ldsB[cur ^ 1], tid);

        const float eq  = esq[nc * 16 + m];   // exact ||e||^2 for this column
        const int  code = nc * 16 + m;

        // K=256 reduction: 8 B fragments, each feeding 2 WMMAs (tiles 0,1)
        v8f acc0 = {0.f, 0.f, 0.f, 0.f, 0.f, 0.f, 0.f, 0.f};
        v8f acc1 = {0.f, 0.f, 0.f, 0.f, 0.f, 0.f, 0.f, 0.f};
        #pragma unroll
        for (int kc = 0; kc < 8; ++kc) {
            const v16bf bf = *(const v16bf*)(&ldsB[cur][m * NC + kc * 32 + kb16]);
            acc0 = __builtin_amdgcn_wmma_f32_16x16x32_bf16(
                false, a0[kc].v, false, bf, (short)0, acc0, false, false);
            acc1 = __builtin_amdgcn_wmma_f32_16x16x32_bf16(
                false, a1[kc].v, false, bf, (short)0, acc1, false, false);
        }

        // dist = ||e||^2 - 2 z.e   (||z||^2 constant per row -> dropped)
        #pragma unroll
        for (int r = 0; r < 8; ++r) {
            const float d0 = __builtin_fmaf(-2.0f, acc0[r], eq);
            const float d1 = __builtin_fmaf(-2.0f, acc1[r], eq);
            if (d0 < best0[r]) { best0[r] = d0; bidx0[r] = code; }
            if (d1 < best1[r]) { best1[r] = d1; bidx1[r] = code; }
        }
        stage_fence();   // staging done + all reads of ldsB[cur] finished
    }

    // ---- Cross-lane argmin within each 16-lane column group ---------------
    #pragma unroll
    for (int r = 0; r < 8; ++r) {
        float v0 = best0[r]; int i0 = bidx0[r];
        float v1 = best1[r]; int i1 = bidx1[r];
        #pragma unroll
        for (int msk = 1; msk <= 8; msk <<= 1) {
            const float ov0 = __shfl_xor(v0, msk, 32);
            const int   oi0 = __shfl_xor(i0, msk, 32);
            const float ov1 = __shfl_xor(v1, msk, 32);
            const int   oi1 = __shfl_xor(i1, msk, 32);
            if (ov0 < v0 || (ov0 == v0 && oi0 < i0)) { v0 = ov0; i0 = oi0; }
            if (ov1 < v1 || (ov1 == v1 && oi1 < i1)) { v1 = ov1; i1 = oi1; }
        }
        bidx0[r] = i0; bidx1[r] = i1;
    }
    // Rows 0..7 replicated in lanes 0-15; rows 8..15 in lanes 16-31.
    // Gather this lane's own tokens' winning indices (token row = m).
    int myInd0 = 0, myInd1 = 0;
    #pragma unroll
    for (int r = 0; r < 8; ++r) {
        const int lo0 = __shfl(bidx0[r], 0,  32);
        const int hi0 = __shfl(bidx0[r], 16, 32);
        const int lo1 = __shfl(bidx1[r], 0,  32);
        const int hi1 = __shfl(bidx1[r], 16, 32);
        if (m == r)     { myInd0 = lo0; myInd1 = lo1; }
        if (m == r + 8) { myInd0 = hi0; myInd1 = hi1; }
    }

    // ---- Epilogue: z_q = emb[ind] scattered to NCHW; loss partials --------
    // lanes 0-15 cover channels 0..127, lanes 16-31 cover 128..255; each half
    // writes 16 consecutive hw -> 64B-coalesced stores, two tiles per pass.
    const int chHalf = hi * 128;
    const size_t rowBase = ((size_t)(b * NC + chHalf)) << 10;
    const float* erow0 = emb + (size_t)myInd0 * NC + chHalf;
    const float* erow1 = emb + (size_t)myInd1 * NC + chHalf;
    float sumSq = 0.f, sumDf = 0.f;
    #pragma unroll 4
    for (int co = 0; co < 128; ++co) {
        const size_t oi0 = rowBase + ((size_t)co << 10) + (size_t)hwA0;
        const float ev0 = erow0[co];
        const float ev1 = erow1[co];
        const float zv0 = zin[oi0];
        const float zv1 = zin[oi0 + 16];
        out[oi0]      = ev0;             // z_q_st == z_q numerically
        out[oi0 + 16] = ev1;
        const float d0 = ev0 - zv0;
        const float d1 = ev1 - zv1;
        sumSq = __builtin_fmaf(d0, d0, sumSq);
        sumSq = __builtin_fmaf(d1, d1, sumSq);
        sumDf += d0 + d1;
    }
    #pragma unroll
    for (int o = 16; o; o >>= 1) {
        sumSq += __shfl_xor(sumSq, o, 32);
        sumDf += __shfl_xor(sumDf, o, 32);
    }
    __syncthreads();   // LDS buffers no longer needed
    if (lane == 0) { rs[wave] = sumSq; rd[wave] = sumDf; }
    __syncthreads();
    if (tid == 0) {
        float S = 0.f, D = 0.f;
        #pragma unroll
        for (int w = 0; w < 8; ++w) { S += rs[w]; D += rd[w]; }
        partials[g] = make_float2(S, D);
    }
}

// ---------------------------------------------------------------------------
// Kernel 3: deterministic final loss reduction over 256 block partials.
// loss = mean((zq - z)^2) + BETA * mean(zq - z)   (source omits the square)
// ---------------------------------------------------------------------------
__global__ void vq_finalize(const float2* __restrict__ partials,
                            float* __restrict__ out_loss) {
    const int t = threadIdx.x;    // 256 threads
    const float2 p = partials[t];
    float sq = p.x;
    float df = p.y;
    #pragma unroll
    for (int o = 16; o; o >>= 1) {
        sq += __shfl_xor(sq, o, 32);
        df += __shfl_xor(df, o, 32);
    }
    __shared__ float rs[8], rd[8];
    if ((t & 31) == 0) { rs[t >> 5] = sq; rd[t >> 5] = df; }
    __syncthreads();
    if (t == 0) {
        float S = 0.f, D = 0.f;
        #pragma unroll
        for (int w = 0; w < 8; ++w) { S += rs[w]; D += rd[w]; }
        const float inv = 1.0f / (float)NELEM;
        out_loss[0] = S * inv + BETA * (D * inv);
    }
}

// ---------------------------------------------------------------------------
extern "C" void kernel_launch(void* const* d_in, const int* in_sizes, int n_in,
                              void* d_out, int out_size, void* d_ws, size_t ws_size,
                              hipStream_t stream) {
    (void)in_sizes; (void)n_in; (void)out_size; (void)ws_size;
    const float* z   = (const float*)d_in[0];   // [64,256,32,32] f32
    const float* emb = (const float*)d_in[1];   // [1024,256] f32
    float* out = (float*)d_out;                  // 16777216 (z_q NCHW) + 1 (loss)

    char* ws = (char*)d_ws;                      // needs ~520 KB
    __bf16* ebf   = (__bf16*)(ws + WS_EBF_OFF);
    float*  esq   = (float*)(ws + WS_ESQ_OFF);
    float2* parts = (float2*)(ws + WS_PART_OFF);

    vq_prep_emb<<<NK, 256, 0, stream>>>(emb, ebf, esq);
    vq_main<<<NTOK / 256, 256, 0, stream>>>(z, emb, ebf, esq, out, parts);
    vq_finalize<<<1, 256, 0, stream>>>(parts, out + NELEM);
}
